// MultiHeadAttention_33148557590552
// MI455X (gfx1250) — matrix operand order repaired and checked by reading, not compiled, not run on hardware
//
#include <hip/hip_runtime.h>
#include <hip/hip_bf16.h>

// ---------------------------------------------------------------------------
// MHA forward for MI455X (gfx1250, wave32, WMMA 16x16x32 f16 -> f32)
//   B=2, S=2048, D=1024, H=16, DK=64
// Kernel 1: double-buffered QKV projection GEMM (templated output layout)
// Kernel 2: flash attention, permlane16 row reductions (no DS in softmax)
// ---------------------------------------------------------------------------

typedef __attribute__((ext_vector_type(16))) _Float16     v16h;
typedef __attribute__((ext_vector_type(4)))  _Float16     v4h;
typedef __attribute__((ext_vector_type(8)))  float        v8f;
typedef __attribute__((ext_vector_type(4)))  float        v4f;
typedef __attribute__((ext_vector_type(8)))  unsigned int v8u;

union Frag { v16h h; v8u u8; unsigned int u[8]; };

constexpr int B_  = 2;
constexpr int S_  = 2048;
constexpr int D_  = 1024;
constexpr int H_  = 16;
constexpr int DK_ = 64;

// A-matrix 16x32 f16 fragment K offsets: lane half h, VGPR i -> contiguous k = h*16 + 2*i (matches B)
__device__ __forceinline__ int ka_off(int i, int h) {
  return h * 16 + 2 * i;
}
// B-matrix fragment: 16 contiguous halves starting at k = h*16

__device__ __forceinline__ v8f wmma_f16(v16h a, v16h b, v8f c) {
  return __builtin_amdgcn_wmma_f32_16x16x32_f16(
      /*neg_a=*/false, a, /*neg_b=*/false, b,
      /*c_mod=*/(short)0, c, /*reuse_a=*/false, /*reuse_b=*/false);
}

// ---- 16-lane row reductions via v_permlane16_b32 (VALU-only, no DS) -------
#if defined(__has_builtin) && __has_builtin(__builtin_amdgcn_permlane16)
__device__ __forceinline__ float perml16(float x, unsigned s0, unsigned s1) {
  unsigned u = __builtin_bit_cast(unsigned, x);
  u = __builtin_amdgcn_permlane16(u, u, s0, s1, false, false);
  return __builtin_bit_cast(float, u);
}
__device__ __forceinline__ float red16_max(float x) {
  x = fmaxf(x, perml16(x, 0x67452301u, 0xEFCDAB89u));  // xor 1
  x = fmaxf(x, perml16(x, 0x54761032u, 0xDCFE98BAu));  // xor 2
  x = fmaxf(x, perml16(x, 0x32107654u, 0xBA98FEDCu));  // xor 4
  x = fmaxf(x, perml16(x, 0xFEDCBA98u, 0x76543210u));  // xor 8
  return x;
}
__device__ __forceinline__ float red16_sum(float x) {
  x += perml16(x, 0x67452301u, 0xEFCDAB89u);
  x += perml16(x, 0x54761032u, 0xDCFE98BAu);
  x += perml16(x, 0x32107654u, 0xBA98FEDCu);
  x += perml16(x, 0xFEDCBA98u, 0x76543210u);
  return x;
}
#else
__device__ __forceinline__ float red16_max(float x) {
  for (int m = 1; m < 16; m <<= 1) x = fmaxf(x, __shfl_xor(x, m, 32));
  return x;
}
__device__ __forceinline__ float red16_sum(float x) {
  for (int m = 1; m < 16; m <<= 1) x += __shfl_xor(x, m, 32);
  return x;
}
#endif

// ---------------------------------------------------------------------------
// Kernel 1: out = (X @ W + bias) * outScale, cast f16, scattered to head layout
//   VT==0 : out[b][h][s][dk]   (Q, K)
//   VT==1 : out[b][h][dk][s]   (V transposed, so attn B-frags are contiguous)
// Tile 128(M) x 128(N) x 32(K), 256 threads = 8 waves (2M x 4N), wave = 64x32.
// Ping-pong LDS double buffer: one barrier per k-step, VMEM overlapped.
// ---------------------------------------------------------------------------
template <int VT>
__global__ __launch_bounds__(256)
void qkv_proj_kernel(const float* __restrict__ X, const float* __restrict__ W,
                     const float* __restrict__ bias, _Float16* __restrict__ out,
                     float outScale)
{
  constexpr int LDT = 40;  // padded row stride (halves); 80B keeps b128 16B-aligned
  __shared__ __align__(16) _Float16 lds[2][2][128 * LDT];  // [buf][A/B][...]

  const int tid  = threadIdx.x;
  const int lane = tid & 31;
  const int w    = tid >> 5;
  const int l15  = lane & 15;
  const int h    = lane >> 4;
  const int wm   = w >> 2;   // 0..1  (64 rows each)
  const int wn   = w & 3;    // 0..3  (32 cols each)

  const int nbase = blockIdx.x * 128;
  const int mbase = blockIdx.y * 128;
  const int bz    = blockIdx.z;

  const float* Xb   = X   + (size_t)bz * S_ * D_;
  _Float16*    outb = out + (size_t)bz * H_ * S_ * DK_;

  // Per-thread staging coordinates (constant across k-steps)
  const int arow0 = tid >> 3;        // +it*32 ; A tile row
  const int acol  = (tid & 7) * 4;   // A tile k-col (float4)
  const int bk0   = tid >> 5;        // +it*8  ; W tile k-row
  const int bn    = (tid & 31) * 4;  // W tile n-col (float4)

  v8f acc[4][2];
  #pragma unroll
  for (int mt = 0; mt < 4; ++mt)
    #pragma unroll
    for (int nt = 0; nt < 2; ++nt)
      acc[mt][nt] = v8f{};

  v4h aReg[4], bReg[4];

  // ---- prologue: stage k-block 0 into buffer 0 ----
  #pragma unroll
  for (int it = 0; it < 4; ++it) {
    v4f x = *(const v4f*)(Xb + (size_t)(mbase + arow0 + it * 32) * D_ + acol);
    aReg[it] = v4h{(_Float16)x.x, (_Float16)x.y, (_Float16)x.z, (_Float16)x.w};
    v4f y = *(const v4f*)(W + (size_t)(bk0 + it * 8) * D_ + nbase + bn);
    bReg[it] = v4h{(_Float16)y.x, (_Float16)y.y, (_Float16)y.z, (_Float16)y.w};
  }
  {
    _Float16* pA = lds[0][0];
    _Float16* pB = lds[0][1];
    #pragma unroll
    for (int it = 0; it < 4; ++it)
      *(v4h*)(pA + (arow0 + it * 32) * LDT + acol) = aReg[it];
    #pragma unroll
    for (int it = 0; it < 4; ++it) {
      pB[(bn + 0) * LDT + bk0 + it * 8] = bReg[it].x;
      pB[(bn + 1) * LDT + bk0 + it * 8] = bReg[it].y;
      pB[(bn + 2) * LDT + bk0 + it * 8] = bReg[it].z;
      pB[(bn + 3) * LDT + bk0 + it * 8] = bReg[it].w;
    }
  }
  __syncthreads();

  int cur = 0;
  for (int kb = 0; kb < D_; kb += 32, cur ^= 1) {
    const bool more = (kb + 32) < D_;

    // ---- issue next tile's global loads (convert to f16 in registers) ----
    if (more) {
      #pragma unroll
      for (int it = 0; it < 4; ++it) {
        v4f x = *(const v4f*)(Xb + (size_t)(mbase + arow0 + it * 32) * D_
                                 + kb + 32 + acol);
        aReg[it] = v4h{(_Float16)x.x, (_Float16)x.y, (_Float16)x.z, (_Float16)x.w};
        v4f y = *(const v4f*)(W + (size_t)(kb + 32 + bk0 + it * 8) * D_ + nbase + bn);
        bReg[it] = v4h{(_Float16)y.x, (_Float16)y.y, (_Float16)y.z, (_Float16)y.w};
      }
    }

    // ---- compute from current buffer ----
    const _Float16* pA = lds[cur][0];
    const _Float16* pB = lds[cur][1];
    Frag af[4], bf[2];
    #pragma unroll
    for (int mt = 0; mt < 4; ++mt) {
      int row = wm * 64 + mt * 16 + l15;
      #pragma unroll
      for (int i = 0; i < 8; ++i)
        af[mt].u[i] = *(const unsigned int*)(pA + row * LDT + ka_off(i, h));
    }
    #pragma unroll
    for (int nt = 0; nt < 2; ++nt) {
      int col = wn * 32 + nt * 16 + l15;
      bf[nt].u8 = *(const v8u*)(pB + col * LDT + h * 16);  // 32B contiguous
    }
    #pragma unroll
    for (int mt = 0; mt < 4; ++mt)
      #pragma unroll
      for (int nt = 0; nt < 2; ++nt)
        acc[mt][nt] = wmma_f16(af[mt].h, bf[nt].h, acc[mt][nt]);

    // ---- store staged registers into the other buffer ----
    if (more) {
      _Float16* qA = lds[cur ^ 1][0];
      _Float16* qB = lds[cur ^ 1][1];
      #pragma unroll
      for (int it = 0; it < 4; ++it)
        *(v4h*)(qA + (arow0 + it * 32) * LDT + acol) = aReg[it];
      #pragma unroll
      for (int it = 0; it < 4; ++it) {
        qB[(bn + 0) * LDT + bk0 + it * 8] = bReg[it].x;
        qB[(bn + 1) * LDT + bk0 + it * 8] = bReg[it].y;
        qB[(bn + 2) * LDT + bk0 + it * 8] = bReg[it].z;
        qB[(bn + 3) * LDT + bk0 + it * 8] = bReg[it].w;
      }
    }
    __syncthreads();
  }

  // ---- epilogue: + bias, * scale, cast f16, scatter (branch-free) ----
  #pragma unroll
  for (int nt = 0; nt < 2; ++nt) {
    int n  = nbase + wn * 32 + nt * 16 + l15;   // global output column
    float bv = bias[n];
    int hd = n >> 6;        // head
    int dk = n & 63;        // within-head dim
    #pragma unroll
    for (int mt = 0; mt < 4; ++mt) {
      #pragma unroll
      for (int r = 0; r < 8; ++r) {
        int   srow = mbase + wm * 64 + mt * 16 + h * 8 + r;
        float v    = (acc[mt][nt][r] + bv) * outScale;
        if (VT)
          outb[((size_t)hd * DK_ + dk) * S_ + srow] = (_Float16)v;
        else
          outb[((size_t)hd * S_ + srow) * DK_ + dk] = (_Float16)v;
      }
    }
  }
}

// ---------------------------------------------------------------------------
// Kernel 2: flash attention. Grid = (S/128, H, B), 256 threads = 8 waves.
// Wave w owns 16 query rows; streams 64-key blocks with online softmax.
// Q/K/V fragments load straight from global f16; P transposed via per-wave LDS.
// Score scale (0.125) was folded into the Q projection.
// ---------------------------------------------------------------------------
__global__ __launch_bounds__(256)
void flash_attn_kernel(const _Float16* __restrict__ Q,
                       const _Float16* __restrict__ K,
                       const _Float16* __restrict__ VT,
                       float* __restrict__ out)
{
  constexpr int PST = 72;  // padded stride (halves) for the 16x64 P slice
  __shared__ __align__(16) _Float16 pst[8][16 * PST];

  const int tid  = threadIdx.x;
  const int lane = tid & 31;
  const int w    = tid >> 5;
  const int l15  = lane & 15;
  const int h    = lane >> 4;

  const int qb = blockIdx.x;   // query block (128 rows)
  const int hd = blockIdx.y;   // head
  const int bz = blockIdx.z;   // batch

  const size_t bh = (size_t)bz * H_ + hd;
  const _Float16* q  = Q  + bh * S_ * DK_;
  const _Float16* k  = K  + bh * S_ * DK_;
  const _Float16* vt = VT + bh * DK_ * S_;   // [dk][s]

  const int qrow = qb * 128 + w * 16 + l15;  // A-frag row for this lane

  // Q fragments (2 K-chunks of 32), kept resident all iterations
  Frag qf[2];
  #pragma unroll
  for (int c = 0; c < 2; ++c)
    #pragma unroll
    for (int i = 0; i < 8; ++i)
      qf[c].u[i] = *(const unsigned int*)(q + (size_t)qrow * DK_ + c * 32 + ka_off(i, h));

  float M[8], L[8];
  v8f o[4];
  #pragma unroll
  for (int r = 0; r < 8; ++r) { M[r] = -1e30f; L[r] = 0.f; }
  #pragma unroll
  for (int dt = 0; dt < 4; ++dt) o[dt] = v8f{};

  constexpr float LOG2E = 1.4426950408889634f;

  for (int kb = 0; kb < S_; kb += 64) {
    // Prefetch next key/value block (global_prefetch_b8); uniform branch
    if (kb + 64 < S_) {
      __builtin_prefetch(k  + (size_t)(kb + 64 + lane * 2) * DK_, 0, 1);
      __builtin_prefetch(vt + (size_t)(lane * 2) * S_ + kb + 64, 0, 1);
    }

    // ---- scores: s[nt] = Q(16x64) . K(16keys x 64)^T ----
    v8f s[4];
    #pragma unroll
    for (int nt = 0; nt < 4; ++nt) {
      s[nt] = v8f{};
      #pragma unroll
      for (int c = 0; c < 2; ++c) {
        Frag kf;
        int key = kb + nt * 16 + l15;  // B-frag column n = key index
        kf.u8 = *(const v8u*)(k + (size_t)key * DK_ + c * 32 + h * 16);  // 32B
        s[nt] = wmma_f16(qf[c].h, kf.h, s[nt]);
      }
    }

    // ---- online softmax (row r lives in 16 lanes of half h at reg r) ----
    float alpha[8];
    #pragma unroll
    for (int r = 0; r < 8; ++r) {
      float rm = red16_max(fmaxf(fmaxf(s[0][r], s[1][r]), fmaxf(s[2][r], s[3][r])));
      float mn = fmaxf(M[r], rm);
      alpha[r] = exp2f((M[r] - mn) * LOG2E);
      M[r] = mn;
    }

    float rsum[8];
    #pragma unroll
    for (int r = 0; r < 8; ++r) rsum[r] = 0.f;
    #pragma unroll
    for (int nt = 0; nt < 4; ++nt)
      #pragma unroll
      for (int r = 0; r < 8; ++r) {
        float p = exp2f((s[nt][r] - M[r]) * LOG2E);
        s[nt][r] = p;
        rsum[r] += p;
      }
    #pragma unroll
    for (int r = 0; r < 8; ++r)
      L[r] = L[r] * alpha[r] + red16_sum(rsum[r]);
    #pragma unroll
    for (int dt = 0; dt < 4; ++dt)
      #pragma unroll
      for (int r = 0; r < 8; ++r)
        o[dt][r] *= alpha[r];

    // ---- transpose P (C-layout -> A-layout) through this wave's LDS slice ----
    _Float16* ps = pst[w];
    #pragma unroll
    for (int nt = 0; nt < 4; ++nt)
      #pragma unroll
      for (int r = 0; r < 8; ++r)
        ps[(h * 8 + r) * PST + nt * 16 + l15] = (_Float16)s[nt][r];
    asm volatile("s_wait_dscnt 0" ::: "memory");  // in-wave DS ordering completes stores

    // ---- o += P(16x64keys) . V(64keys x 64dk), V-frags from transposed VT ----
    #pragma unroll
    for (int c = 0; c < 2; ++c) {
      Frag pf;
      #pragma unroll
      for (int i = 0; i < 8; ++i)
        pf.u[i] = *(const unsigned int*)(ps + l15 * PST + c * 32 + ka_off(i, h));
      #pragma unroll
      for (int dt = 0; dt < 4; ++dt) {
        Frag vf;
        vf.u8 = *(const v8u*)(vt + (size_t)(dt * 16 + l15) * S_
                                 + kb + c * 32 + h * 16);   // 32B contiguous
        o[dt] = wmma_f16(pf.h, vf.h, o[dt]);
      }
    }
  }

  // ---- normalize and write ctx back to [B, S, D] f32 ----
  float inv[8];
  #pragma unroll
  for (int r = 0; r < 8; ++r) inv[r] = 1.0f / L[r];
  #pragma unroll
  for (int dt = 0; dt < 4; ++dt) {
    #pragma unroll
    for (int r = 0; r < 8; ++r) {
      int srow = qb * 128 + w * 16 + h * 8 + r;
      out[((size_t)bz * S_ + srow) * D_ + hd * DK_ + dt * 16 + l15] = o[dt][r] * inv[r];
    }
  }
}

// ---------------------------------------------------------------------------
extern "C" void kernel_launch(void* const* d_in, const int* in_sizes, int n_in,
                              void* d_out, int out_size, void* d_ws, size_t ws_size,
                              hipStream_t stream) {
  const float* query = (const float*)d_in[0];
  const float* key_  = (const float*)d_in[1];
  const float* value = (const float*)d_in[2];
  const float* Wq    = (const float*)d_in[3];
  const float* bq    = (const float*)d_in[4];
  const float* Wk    = (const float*)d_in[5];
  const float* bk    = (const float*)d_in[6];
  const float* Wv    = (const float*)d_in[7];
  const float* bv    = (const float*)d_in[8];

  const size_t perMat = (size_t)B_ * H_ * S_ * DK_;   // 4M halves = 8 MB
  _Float16* q16 = (_Float16*)d_ws;
  _Float16* k16 = q16 + perMat;
  _Float16* v16 = k16 + perMat;   // stored transposed [B,H,DK,S]

  dim3 gProj(D_ / 128, S_ / 128, B_);   // (8, 16, 2)
  // Fold 1/sqrt(DK) = 0.125 into the Q projection.
  qkv_proj_kernel<0><<<gProj, 256, 0, stream>>>(query, Wq, bq, q16, 0.125f);
  qkv_proj_kernel<0><<<gProj, 256, 0, stream>>>(key_,  Wk, bk, k16, 1.0f);
  qkv_proj_kernel<1><<<gProj, 256, 0, stream>>>(value, Wv, bv, v16, 1.0f);

  dim3 gAttn(S_ / 128, H_, B_);         // (16, 16, 2)
  flash_attn_kernel<<<gAttn, 256, 0, stream>>>(q16, k16, v16, (float*)d_out);
}
